// Denoising_Unet_39857296507357
// MI455X (gfx1250) — compile-verified, matrix operand order checked by reading, not run on hardware
//
#include <hip/hip_runtime.h>
#include <hip/hip_bf16.h>
#include <float.h>
#include <math.h>

#define NN   10000
#define EE   160000
#define HIDD 512
#define HEADS 8
#define DHH   64
#define KC   64   // K chunk staged in LDS per iteration

typedef __attribute__((ext_vector_type(16))) __bf16 v16bf;
typedef __attribute__((ext_vector_type(8)))  __bf16 v8bf;
typedef __attribute__((ext_vector_type(8)))  float  v8f;
typedef __attribute__((ext_vector_type(4)))  unsigned int v4u;
typedef __attribute__((ext_vector_type(4)))  int v4i;

#if __has_builtin(__builtin_amdgcn_global_load_async_to_lds_b128) && \
    __has_builtin(__builtin_amdgcn_s_wait_asynccnt)
#define DU_ASYNC_LDS 1
typedef __attribute__((address_space(1))) v4i as1_v4i;   // global memory int4
typedef __attribute__((address_space(3))) v4i as3_v4i;   // LDS int4
#else
#define DU_ASYNC_LDS 0
#endif

static __device__ __forceinline__ void atomicMaxF(float* addr, float v) {
    // standard monotonic-bit-pattern trick: works for mixed-sign floats
    if (v >= 0.0f) atomicMax((int*)addr, __float_as_int(v));
    else           atomicMin((unsigned int*)addr, __float_as_uint(v));
}

// ---------------------------------------------------------------- elementwise
__global__ void du_fill_kernel(float* __restrict__ p, float v, int n) {
    int t = blockIdx.x * blockDim.x + threadIdx.x;
    if (t < n) p[t] = v;
}

__global__ void du_concat_kernel(const float* __restrict__ x, const float* __restrict__ te,
                                 float* __restrict__ h, int n) {
    int t = blockIdx.x * blockDim.x + threadIdx.x;
    if (t >= n) return;
    int node = t >> 10, c = t & 1023;
    h[t] = (c < HIDD) ? x[(size_t)node * HIDD + c] : te[(size_t)node * HIDD + (c - HIDD)];
}

__global__ void du_cvt_bf16_kernel(const float* __restrict__ s, __bf16* __restrict__ d, int n) {
    int t = blockIdx.x * blockDim.x + threadIdx.x;
    if (t < n) d[t] = (__bf16)s[t];
}

// W is [K x Nout] row-major fp32 -> Wt is [Nout x K] row-major bf16
__global__ void du_transpose_cvt_kernel(const float* __restrict__ W, __bf16* __restrict__ Wt,
                                        int K, int Nout) {
    int t = blockIdx.x * blockDim.x + threadIdx.x;
    if (t >= K * Nout) return;
    int k = t / Nout, n = t - k * Nout;
    Wt[(size_t)n * K + k] = (__bf16)W[t];
}

// ---------------------------------------------------------------- WMMA GEMM
// C[M x Nout] = A[M x K](bf16) * Bt[Nout x K]^T(bf16) + bias, f32 accumulate.
// Block: 256 threads = 8 wave32 waves. Block tile 128(M) x 64(N), K staged KC=64
// at a time through LDS (async-copy path on CDNA5), 8 WMMAs per barrier.
__global__ __launch_bounds__(256)
void du_gemm_bf16_wmma(const __bf16* __restrict__ A, const __bf16* __restrict__ Bt,
                       const float* __restrict__ bias, float* __restrict__ C,
                       int M, int K, int Nout, int act) {
    __shared__ __bf16 As[128][KC];
    __shared__ __bf16 Bs[64][KC];

    const int tid   = threadIdx.x;
    const int wave  = tid >> 5;        // 0..7 (wave32)
    const int lane  = tid & 31;
    const int lr    = lane & 15;
    const int khalf = (lane >> 4) * 8; // 0 or 8
    const int rowBase = blockIdx.x * 128;
    const int colBase = blockIdx.y * 64;

    v8f acc[4];
    const v8f vzero = {0.f, 0.f, 0.f, 0.f, 0.f, 0.f, 0.f, 0.f};
#pragma unroll
    for (int s = 0; s < 4; ++s) acc[s] = vzero;

    for (int kc = 0; kc < K; kc += KC) {
        // stage A tile 128 x KC : 1024 chunks of 8 bf16 (16B), 4 per thread.
        // Rows past M are clamped (their outputs are never stored).
#pragma unroll
        for (int i = 0; i < 4; ++i) {
            int chunk = tid + i * 256;
            int r  = chunk >> 3;
            int c8 = (chunk & 7) * 8;
            int grow = rowBase + r;
            if (grow >= M) grow = M - 1;
            const __bf16* gp = A + (size_t)grow * K + kc + c8;
            __bf16* lp = &As[r][c8];
#if DU_ASYNC_LDS
            __builtin_amdgcn_global_load_async_to_lds_b128(
                (as1_v4i*)gp, (as3_v4i*)lp, 0, 0);
#else
            *(v4u*)lp = *(const v4u*)gp;
#endif
        }
        // stage Bt tile 64 x KC : 512 chunks, 2 per thread
#pragma unroll
        for (int i = 0; i < 2; ++i) {
            int chunk = tid + i * 256;
            int r  = chunk >> 3;
            int c8 = (chunk & 7) * 8;
            const __bf16* gp = Bt + (size_t)(colBase + r) * K + kc + c8;
            __bf16* lp = &Bs[r][c8];
#if DU_ASYNC_LDS
            __builtin_amdgcn_global_load_async_to_lds_b128(
                (as1_v4i*)gp, (as3_v4i*)lp, 0, 0);
#else
            *(v4u*)lp = *(const v4u*)gp;
#endif
        }
#if DU_ASYNC_LDS
        __builtin_amdgcn_s_wait_asynccnt(0);   // own wave's async copies done
#endif
        __syncthreads();                       // all waves' copies visible

        if (kc + KC < K) {                     // prefetch next A chunk (global_prefetch_b8)
            int pr = rowBase + (tid >> 3);
            if (pr >= M) pr = M - 1;
            __builtin_prefetch((const void*)(A + (size_t)pr * K + kc + KC), 0, 1);
        }

        const int ar = wave * 16 + lr;
#pragma unroll
        for (int kk = 0; kk < KC; kk += 32) {
            v8bf a0 = *(const v8bf*)&As[ar][kk + khalf];
            v8bf a1 = *(const v8bf*)&As[ar][kk + 16 + khalf];
            v16bf av = __builtin_shufflevector(a0, a1,
                0,1,2,3,4,5,6,7,8,9,10,11,12,13,14,15);
#pragma unroll
            for (int s = 0; s < 4; ++s) {
                const int br = s * 16 + lr;
                v8bf b0 = *(const v8bf*)&Bs[br][kk + khalf];
                v8bf b1 = *(const v8bf*)&Bs[br][kk + 16 + khalf];
                v16bf bv = __builtin_shufflevector(b0, b1,
                    0,1,2,3,4,5,6,7,8,9,10,11,12,13,14,15);
                acc[s] = __builtin_amdgcn_wmma_f32_16x16x32_bf16(
                    false, av, false, bv, (short)0, acc[s], false, false);
            }
        }
        __syncthreads();
    }

    // epilogue: D layout -> lanes 0-15: M=j, lanes 16-31: M=8+j; N = lane&15
    const int rowOff = rowBase + wave * 16 + ((lane >> 4) ? 8 : 0);
#pragma unroll
    for (int s = 0; s < 4; ++s) {
        int col = colBase + s * 16 + lr;
        float bv = bias ? bias[col] : 0.0f;
#pragma unroll
        for (int j = 0; j < 8; ++j) {
            int row = rowOff + j;
            if (row < M) {
                float v = acc[s][j] + bv;
                if (act == 1) v = fmaxf(v, 0.0f);
                C[(size_t)row * Nout + col] = v;
            }
        }
    }
}

// ---------------------------------------------------------------- edge phase
__global__ void du_edge_logits_kernel(const float* __restrict__ fs, const float* __restrict__ fd,
                                      const float* __restrict__ attn,
                                      const int* __restrict__ src, const int* __restrict__ dst,
                                      float* __restrict__ logits, float* __restrict__ m, int E) {
    int t = blockIdx.x * blockDim.x + threadIdx.x;
    if (t >= E * HEADS) return;
    int e = t >> 3, h = t & 7;
    int s = src[e], d = dst[e];
    const float4* fsr = (const float4*)(fs + (size_t)s * HIDD + h * DHH);
    const float4* fdr = (const float4*)(fd + (size_t)d * HIDD + h * DHH);
    const float4* ar  = (const float4*)(attn + h * DHH);
    float acc = 0.0f;
#pragma unroll
    for (int i = 0; i < DHH / 4; ++i) {
        float4 a4 = fsr[i], b4 = fdr[i], w4 = ar[i];
        float v;
        v = a4.x + b4.x; v = v > 0.f ? v : 0.2f * v; acc += v * w4.x;
        v = a4.y + b4.y; v = v > 0.f ? v : 0.2f * v; acc += v * w4.y;
        v = a4.z + b4.z; v = v > 0.f ? v : 0.2f * v; acc += v * w4.z;
        v = a4.w + b4.w; v = v > 0.f ? v : 0.2f * v; acc += v * w4.w;
    }
    logits[t] = acc;
    atomicMaxF(&m[(size_t)d * HEADS + h], acc);
}

__global__ void du_edge_exp_kernel(float* __restrict__ logits, const float* __restrict__ m,
                                   const int* __restrict__ dst, float* __restrict__ denom, int E) {
    int t = blockIdx.x * blockDim.x + threadIdx.x;
    if (t >= E * HEADS) return;
    int e = t >> 3, h = t & 7;
    int d = dst[e];
    float ex = expf(logits[t] - m[(size_t)d * HEADS + h]);
    logits[t] = ex;
    atomicAdd(&denom[(size_t)d * HEADS + h], ex);
}

__global__ void du_edge_agg_kernel(const float* __restrict__ ex, const float* __restrict__ denom,
                                   const float* __restrict__ fs,
                                   const int* __restrict__ src, const int* __restrict__ dst,
                                   float* __restrict__ agg, int E) {
    int t = blockIdx.x * blockDim.x + threadIdx.x;
    if (t >= E * HIDD) return;
    int e = t >> 9, c = t & (HIDD - 1), h = c >> 6;
    int s = src[e], d = dst[e];
    float alpha = ex[(size_t)e * HEADS + h] / fmaxf(denom[(size_t)d * HEADS + h], 1e-9f);
    atomicAdd(&agg[(size_t)d * HIDD + c], alpha * fs[(size_t)s * HIDD + c]);
}

__global__ void du_gelu_res_kernel(float* __restrict__ agg, const float* __restrict__ hin,
                                   int use_res, int n) {
    int t = blockIdx.x * blockDim.x + threadIdx.x;
    if (t >= n) return;
    float v = agg[t];
    v = 0.5f * v * (1.0f + erff(v * 0.70710678118654752f));   // exact gelu
    if (use_res) v += hin[t];
    agg[t] = v;
}

// one wave32 per node
__global__ void du_layernorm_kernel(const float* __restrict__ x, const float* __restrict__ g,
                                    const float* __restrict__ b, float* __restrict__ y, int n) {
    int wave = threadIdx.x >> 5, lane = threadIdx.x & 31;
    int node = blockIdx.x * 8 + wave;
    if (node >= n) return;
    const float* xr = x + (size_t)node * HIDD;
    float v[HIDD / 32];
    float s = 0.0f;
#pragma unroll
    for (int i = 0; i < HIDD / 32; ++i) { v[i] = xr[lane + i * 32]; s += v[i]; }
#pragma unroll
    for (int o = 16; o > 0; o >>= 1) s += __shfl_xor(s, o, 32);
    float mu = s * (1.0f / HIDD);
    float q = 0.0f;
#pragma unroll
    for (int i = 0; i < HIDD / 32; ++i) { float dd = v[i] - mu; q += dd * dd; }
#pragma unroll
    for (int o = 16; o > 0; o >>= 1) q += __shfl_xor(q, o, 32);
    float r = rsqrtf(q * (1.0f / HIDD) + 1e-5f);
    float* yr = y + (size_t)node * HIDD;
#pragma unroll
    for (int i = 0; i < HIDD / 32; ++i) {
        int c = lane + i * 32;
        yr[c] = (v[i] - mu) * r * g[c] + b[c];
    }
}

// ---------------------------------------------------------------- launch
static inline int cdiv(int a, int b) { return (a + b - 1) / b; }

extern "C" void kernel_launch(void* const* d_in, const int* in_sizes, int n_in,
                              void* d_out, int out_size, void* d_ws, size_t ws_size,
                              hipStream_t stream) {
    const float* x_t   = (const float*)d_in[0];
    const float* temb  = (const float*)d_in[1];
    const int*   src   = (const int*)d_in[2];
    const int*   dst   = (const int*)d_in[3];
    const float* Wsrc[3] = {(const float*)d_in[4],  (const float*)d_in[5],  (const float*)d_in[6]};
    const float* bsrc[3] = {(const float*)d_in[7],  (const float*)d_in[8],  (const float*)d_in[9]};
    const float* Wdst[3] = {(const float*)d_in[10], (const float*)d_in[11], (const float*)d_in[12]};
    const float* bdst[3] = {(const float*)d_in[13], (const float*)d_in[14], (const float*)d_in[15]};
    const float* attn[3] = {(const float*)d_in[16], (const float*)d_in[17], (const float*)d_in[18]};
    const float* lng[3]  = {(const float*)d_in[19], (const float*)d_in[20], (const float*)d_in[21]};
    const float* lnb[3]  = {(const float*)d_in[22], (const float*)d_in[23], (const float*)d_in[24]};
    const float* Wo1 = (const float*)d_in[25];
    const float* bo1 = (const float*)d_in[26];
    const float* Wo2 = (const float*)d_in[27];
    const float* bo2 = (const float*)d_in[28];

    // workspace carve-up
    char* wp = (char*)d_ws;
    auto alloc = [&](size_t bytes) -> void* {
        void* r = (void*)wp;
        wp += (bytes + 255) & ~(size_t)255;
        return r;
    };
    const size_t NH = (size_t)NN * HIDD;
    float*  hcat  = (float*)alloc((size_t)NN * 1024 * sizeof(float));
    float*  h1    = (float*)alloc(NH * sizeof(float));
    float*  h2    = (float*)alloc(NH * sizeof(float));
    __bf16* hbf   = (__bf16*)alloc((size_t)NN * 1024 * sizeof(__bf16));
    __bf16* Wt    = (__bf16*)alloc((size_t)1024 * 512 * sizeof(__bf16));
    float*  fs    = (float*)alloc(NH * sizeof(float));
    float*  fd    = (float*)alloc(NH * sizeof(float));
    float*  logit = (float*)alloc((size_t)EE * HEADS * sizeof(float));
    float*  mbuf  = (float*)alloc((size_t)NN * HEADS * sizeof(float));
    float*  dbuf  = (float*)alloc((size_t)NN * HEADS * sizeof(float));
    float*  agg   = (float*)alloc(NH * sizeof(float));
    float*  tmp   = (float*)alloc(NH * sizeof(float));
    __bf16* tmpbf = (__bf16*)alloc(NH * sizeof(__bf16));
    (void)ws_size; (void)in_sizes; (void)n_in; (void)out_size;

    const int TB = 256;

    // h0 = concat(x_t, time_embed)
    {
        int n = NN * 1024;
        du_concat_kernel<<<cdiv(n, TB), TB, 0, stream>>>(x_t, temb, hcat, n);
    }

    float* hfin = (float*)d_out + NH;   // second tuple output: final h
    const float* hin = hcat;
    int Kin = 1024;

    for (int l = 0; l < 3; ++l) {
        float* hout = (l == 0) ? h1 : (l == 1) ? h2 : hfin;

        // A matrix in bf16
        {
            int n = NN * Kin;
            du_cvt_bf16_kernel<<<cdiv(n, TB), TB, 0, stream>>>(hin, hbf, n);
        }
        dim3 gg(cdiv(NN, 128), HIDD / 64);
        // fs = h @ Wsrc + bsrc
        du_transpose_cvt_kernel<<<cdiv(Kin * HIDD, TB), TB, 0, stream>>>(Wsrc[l], Wt, Kin, HIDD);
        du_gemm_bf16_wmma<<<gg, TB, 0, stream>>>(hbf, Wt, bsrc[l], fs, NN, Kin, HIDD, 0);
        // fd = h @ Wdst + bdst
        du_transpose_cvt_kernel<<<cdiv(Kin * HIDD, TB), TB, 0, stream>>>(Wdst[l], Wt, Kin, HIDD);
        du_gemm_bf16_wmma<<<gg, TB, 0, stream>>>(hbf, Wt, bdst[l], fd, NN, Kin, HIDD, 0);

        // softmax state init
        du_fill_kernel<<<cdiv(NN * HEADS, TB), TB, 0, stream>>>(mbuf, -FLT_MAX, NN * HEADS);
        du_fill_kernel<<<cdiv(NN * HEADS, TB), TB, 0, stream>>>(dbuf, 0.0f, NN * HEADS);
        du_fill_kernel<<<cdiv((int)NH, TB), TB, 0, stream>>>(agg, 0.0f, (int)NH);

        // edge phase
        du_edge_logits_kernel<<<cdiv(EE * HEADS, TB), TB, 0, stream>>>(fs, fd, attn[l], src, dst,
                                                                       logit, mbuf, EE);
        du_edge_exp_kernel<<<cdiv(EE * HEADS, TB), TB, 0, stream>>>(logit, mbuf, dst, dbuf, EE);
        du_edge_agg_kernel<<<cdiv(EE * HIDD, TB), TB, 0, stream>>>(logit, dbuf, fs, src, dst,
                                                                   agg, EE);

        // gelu + residual + layernorm
        du_gelu_res_kernel<<<cdiv((int)NH, TB), TB, 0, stream>>>(agg, hin, (l > 0) ? 1 : 0, (int)NH);
        du_layernorm_kernel<<<cdiv(NN, 8), TB, 0, stream>>>(agg, lng[l], lnb[l], hout, NN);

        hin = hout;
        Kin = HIDD;
    }

    // output MLP: out = relu(h @ Wo1 + bo1) @ Wo2 + bo2
    {
        dim3 gg(cdiv(NN, 128), HIDD / 64);
        du_cvt_bf16_kernel<<<cdiv((int)NH, TB), TB, 0, stream>>>(hin, hbf, (int)NH);
        du_transpose_cvt_kernel<<<cdiv(HIDD * HIDD, TB), TB, 0, stream>>>(Wo1, Wt, HIDD, HIDD);
        du_gemm_bf16_wmma<<<gg, TB, 0, stream>>>(hbf, Wt, bo1, tmp, NN, HIDD, HIDD, 1);
        du_cvt_bf16_kernel<<<cdiv((int)NH, TB), TB, 0, stream>>>(tmp, tmpbf, (int)NH);
        du_transpose_cvt_kernel<<<cdiv(HIDD * HIDD, TB), TB, 0, stream>>>(Wo2, Wt, HIDD, HIDD);
        du_gemm_bf16_wmma<<<gg, TB, 0, stream>>>(tmpbf, Wt, bo2, (float*)d_out, NN, HIDD, HIDD, 0);
    }
}
